// MPGLNet_16836271801143
// MI455X (gfx1250) — matrix operand order repaired and checked by reading, not compiled, not run on hardware
//
#include <hip/hip_runtime.h>

#define BB 2
#define CC 512
#define NN 2048
#define MM 512
#define SS (NN + MM)
#define KK 16
#define PH 64
#define AH 1024
#define BN_EPS 1e-5f

typedef __attribute__((ext_vector_type(16))) __bf16 v16bf;
typedef __attribute__((ext_vector_type(8))) float v8f;

__device__ __forceinline__ unsigned short f2bf(float f) {
  unsigned u = __float_as_uint(f);
  u += 0x7FFFu + ((u >> 16) & 1u);   // round-to-nearest-even
  return (unsigned short)(u >> 16);
}
__device__ __forceinline__ float bf2f(unsigned short h) {
  return __uint_as_float(((unsigned)h) << 16);
}

// A-matrix fragment (16x32 bf16). Lane (h = lane>>4, m = lane&15) holds row m:
//   elems 0..7  -> K = c0 + h*8 + e        (16 contiguous bytes)
//   elems 8..15 -> K = c0 + 16 + h*8 + e-8 (16 contiguous bytes, +32B away)
// p points at &W[row*Cin + c0 + h*8] in a row-major bf16 weight matrix.
__device__ __forceinline__ v16bf load_a_frag(const unsigned short* p) {
  union { v16bf v; uint4 q[2]; } u;
  u.q[0] = *reinterpret_cast<const uint4*>(p);
  u.q[1] = *reinterpret_cast<const uint4*>(p + 16);
  return u.v;
}

// B-matrix fragment (32x16 bf16). Lane (h, n = lane&15) holds column n:
//   elems 0..7  -> K = c0 + h*8 + e        (16 contiguous bytes)
//   elems 8..15 -> K = c0 + 16 + h*8 + e-8 (16 contiguous bytes, +32B away)
// p points at &x_t[n][c0 + h*8] where x_t is [16][Cin] (channel-contiguous).
__device__ __forceinline__ v16bf load_b_frag(const unsigned short* p) {
  union { v16bf v; uint4 q[2]; } u;
  u.q[0] = *reinterpret_cast<const uint4*>(p);
  u.q[1] = *reinterpret_cast<const uint4*>(p + 16);
  return u.v;
}

// ---------------------------------------------------------------- pack ------
__global__ __launch_bounds__(256) void pack_kernel(
    const float* __restrict__ pw2, const float* __restrict__ aw1,
    const float* __restrict__ aw2, const float* __restrict__ ag,
    const float* __restrict__ ab, const float* __restrict__ am,
    const float* __restrict__ av, unsigned short* __restrict__ w2p,
    unsigned short* __restrict__ a1p, unsigned short* __restrict__ a2p,
    float* __restrict__ ainv, float* __restrict__ ashift) {
  int i = blockIdx.x * 256 + threadIdx.x;
  if (i < CC * PH) w2p[i] = f2bf(pw2[i]);
  if (i < AH * CC) { a1p[i] = f2bf(aw1[i]); a2p[i] = f2bf(aw2[i]); }
  if (i < AH) {
    float inv = ag[i] * rsqrtf(av[i] + BN_EPS);
    ainv[i] = inv;
    ashift[i] = ab[i] - am[i] * inv;
  }
}

// ---------------------------------------------------------------- knn -------
__global__ __launch_bounds__(256) void knn_kernel(
    const float* __restrict__ pcd, const float* __restrict__ pcdb,
    int* __restrict__ idx_out) {
  int g = blockIdx.x * 256 + threadIdx.x;
  if (g >= BB * NN) return;
  int b = g / NN, q = g - b * NN;
  const float* pq = pcd + (size_t)b * 3 * NN;
  const float* pb_ = pcdb + (size_t)b * 3 * MM;
  float qx = pq[q], qy = pq[NN + q], qz = pq[2 * NN + q];
  float qn = qx * qx + qy * qy + qz * qz;
  float dl[KK];
  int il[KK];
#pragma unroll
  for (int j = 0; j < KK; ++j) { dl[j] = 3.0e38f; il[j] = 0; }
  for (int s = 0; s < SS; ++s) {
    float sx, sy, sz;
    if (s < NN) { sx = pq[s]; sy = pq[NN + s]; sz = pq[2 * NN + s]; }
    else { int t = s - NN; sx = pb_[t]; sy = pb_[MM + t]; sz = pb_[2 * MM + t]; }
    float d = qn + sx * sx + sy * sy + sz * sz
            - 2.0f * (qx * sx + qy * sy + qz * sz);
    if (d < dl[KK - 1]) {
      dl[KK - 1] = d; il[KK - 1] = s;
#pragma unroll
      for (int j = KK - 1; j > 0; --j) {      // strict < keeps tie order stable
        if (dl[j] < dl[j - 1]) {
          float td = dl[j]; dl[j] = dl[j - 1]; dl[j - 1] = td;
          int ti = il[j]; il[j] = il[j - 1]; il[j - 1] = ti;
        }
      }
    }
  }
#pragma unroll
  for (int j = 0; j < KK; ++j) idx_out[(size_t)g * KK + j] = il[j];
}

// ---------------------------------------------------------------- fused -----
__global__ __launch_bounds__(256, 1) void fused_kernel(
    const float* __restrict__ pcd, const float* __restrict__ feat,
    const float* __restrict__ pcdb, const float* __restrict__ featb,
    const float* __restrict__ pw1, const float* __restrict__ pb1,
    const float* __restrict__ pg, const float* __restrict__ pbeta,
    const float* __restrict__ pmean, const float* __restrict__ pvar,
    const float* __restrict__ pb2, const float* __restrict__ ab1,
    const float* __restrict__ ab2, const unsigned short* __restrict__ w2p,
    const unsigned short* __restrict__ a1p, const unsigned short* __restrict__ a2p,
    const float* __restrict__ ainv, const float* __restrict__ ashift,
    const int* __restrict__ knn, float* __restrict__ out) {
  __shared__ int idx_s[KK];
  __shared__ float prel[3][KK];
  __shared__ __align__(16) unsigned short h_t[KK][PH];       // pos hidden, 2KB
  __shared__ __align__(16) unsigned short x_t[KK][CC];       // attn input, 16KB
  __shared__ __align__(16) unsigned short gfpe_t[KK][CC];    // gf+pe, 16KB
  __shared__ __align__(16) unsigned short a_t[KK][CC];       // attn hidden half, 16KB

  const int tid = threadIdx.x;
  const int wave = tid >> 5;
  const int lane = tid & 31;
  const int hh = lane >> 4;   // lane half
  const int ln = lane & 15;   // row (A) / column (B,C,D) within half
  const int bq = blockIdx.x;
  const int b = bq / NN;
  const int q = bq - b * NN;

  // ---- phase 0: neighbor indices + pos_rel ----
  if (tid < KK) idx_s[tid] = knn[(size_t)bq * KK + tid];
  __syncthreads();
  if (tid < 3 * KK) {
    int d = tid >> 4, k = tid & 15;
    int i = idx_s[k];
    const float* row = pcd + ((size_t)b * 3 + d) * NN;
    float gp = (i < NN) ? row[i] : pcdb[((size_t)b * 3 + d) * MM + (i - NN)];
    prel[d][k] = row[q] - gp;
  }
  __syncthreads();

  // ---- phase 1: pos layer1 (3 -> 64) + BN + ReLU (scalar, tiny) ----
  {
    int n = tid & 15;
    int c0 = (tid >> 4) * 4;
    float r0 = prel[0][n], r1 = prel[1][n], r2 = prel[2][n];
#pragma unroll
    for (int j = 0; j < 4; ++j) {
      int c = c0 + j;
      float v = pw1[c * 3 + 0] * r0 + pw1[c * 3 + 1] * r1 +
                pw1[c * 3 + 2] * r2 + pb1[c];
      float inv = pg[c] * rsqrtf(pvar[c] + BN_EPS);
      v = v * inv + (pbeta[c] - pmean[c] * inv);
      h_t[n][c] = f2bf(fmaxf(v, 0.0f));
    }
  }
  __syncthreads();

  // ---- phase 2: pos conv2 (64 -> 512) WMMA; gather; build x and gf+pe ----
  // kk-outer / tile-inner: each B fragment is loaded once and feeds 4 WMMAs.
  {
    v8f acc2[4] = {};
#pragma unroll
    for (int kk = 0; kk < PH / 32; ++kk) {
      int c0 = kk * 32;
      v16bf bm = load_b_frag(&h_t[ln][c0 + hh * 8]);
#pragma unroll
      for (int j = 0; j < 4; ++j) {
        int tile = wave + 8 * j;
        v16bf a = load_a_frag(w2p + (size_t)(tile * 16 + ln) * PH + c0 + hh * 8);
        acc2[j] = __builtin_amdgcn_wmma_f32_16x16x32_bf16(
            false, a, false, bm, (short)0, acc2[j], false, false);
      }
    }
    int i = idx_s[ln];
    const float* gbase;
    int gs;
    if (i < NN) { gbase = feat + (size_t)b * CC * NN + i; gs = NN; }
    else        { gbase = featb + (size_t)b * CC * MM + (i - NN); gs = MM; }
    const float* fqb = feat + (size_t)b * CC * NN + q;
#pragma unroll
    for (int j = 0; j < 4; ++j) {
      int cbase = (wave + 8 * j) * 16 + hh * 8;
      float xv[8], gv[8];
#pragma unroll
      for (int r = 0; r < 8; ++r) {
        int m = cbase + r;
        float pe = acc2[j][r] + pb2[m];
        float gf = gbase[(size_t)m * gs];
        float fq = fqb[(size_t)m * NN];
        xv[r] = fq - gf + pe;   // qk_rel + pos_embedding
        gv[r] = gf + pe;        // group_feat + pos_embedding
      }
#pragma unroll
      for (int r = 0; r < 8; r += 2) {
        *(unsigned*)&x_t[ln][cbase + r] =
            (unsigned)f2bf(xv[r]) | ((unsigned)f2bf(xv[r + 1]) << 16);
        *(unsigned*)&gfpe_t[ln][cbase + r] =
            (unsigned)f2bf(gv[r]) | ((unsigned)f2bf(gv[r + 1]) << 16);
      }
    }
  }
  __syncthreads();

  // ---- phases 3+4: attn conv1 (512->1024) and conv2 (1024->512), in two
  //      512-channel halves of the hidden dim; conv2 partials in registers.
  //      kk-outer / tile-inner throughout: 1 B load pair -> 4 WMMAs. ----
  v8f accs[4] = {};
  for (int half = 0; half < 2; ++half) {
    // attn conv1 for hidden channels [half*512, half*512+512)
    {
      v8f acc1[4] = {};
#pragma unroll 4
      for (int kk = 0; kk < CC / 32; ++kk) {
        int c0 = kk * 32;
        v16bf bm = load_b_frag(&x_t[ln][c0 + hh * 8]);
#pragma unroll
        for (int t = 0; t < 4; ++t) {
          int th = wave + 8 * t;
          v16bf a = load_a_frag(a1p + (size_t)(half * 512 + th * 16 + ln) * CC +
                                c0 + hh * 8);
          acc1[t] = __builtin_amdgcn_wmma_f32_16x16x32_bf16(
              false, a, false, bm, (short)0, acc1[t], false, false);
        }
      }
#pragma unroll
      for (int t = 0; t < 4; ++t) {
        int th = wave + 8 * t;
        int gm = half * 512 + th * 16 + hh * 8;
        int cb = th * 16 + hh * 8;
        float av_[8];
#pragma unroll
        for (int r = 0; r < 8; ++r) {
          float v = acc1[t][r] + ab1[gm + r];
          av_[r] = fmaxf(v * ainv[gm + r] + ashift[gm + r], 0.0f);
        }
#pragma unroll
        for (int r = 0; r < 8; r += 2)
          *(unsigned*)&a_t[ln][cb + r] =
              (unsigned)f2bf(av_[r]) | ((unsigned)f2bf(av_[r + 1]) << 16);
      }
    }
    __syncthreads();
    // attn conv2 partial accumulation over this hidden half
#pragma unroll 4
    for (int kk = 0; kk < 16; ++kk) {
      int c0 = kk * 32;
      v16bf bm = load_b_frag(&a_t[ln][c0 + hh * 8]);
#pragma unroll
      for (int j = 0; j < 4; ++j) {
        int ot = wave + 8 * j;
        v16bf a = load_a_frag(a2p + (size_t)(ot * 16 + ln) * AH + half * 512 +
                              c0 + hh * 8);
        accs[j] = __builtin_amdgcn_wmma_f32_16x16x32_bf16(
            false, a, false, bm, (short)0, accs[j], false, false);
      }
    }
    __syncthreads();
  }

  // ---- phase 5: softmax over K (across 16 lanes) + weighted sum + store ----
#pragma unroll
  for (int j = 0; j < 4; ++j) {
    int ot = wave + 8 * j;
    int cbase = ot * 16 + hh * 8;
    uint4 gq = *(const uint4*)&gfpe_t[ln][cbase];
    unsigned gw[4] = {gq.x, gq.y, gq.z, gq.w};
    float osum[8];
#pragma unroll
    for (int r = 0; r < 8; ++r) {
      float v = accs[j][r] + ab2[cbase + r];
      float mx = v;
      mx = fmaxf(mx, __shfl_xor(mx, 1, 32));
      mx = fmaxf(mx, __shfl_xor(mx, 2, 32));
      mx = fmaxf(mx, __shfl_xor(mx, 4, 32));
      mx = fmaxf(mx, __shfl_xor(mx, 8, 32));
      float e = __expf(v - mx);
      float s = e;
      s += __shfl_xor(s, 1, 32);
      s += __shfl_xor(s, 2, 32);
      s += __shfl_xor(s, 4, 32);
      s += __shfl_xor(s, 8, 32);
      unsigned gpk = gw[r >> 1];
      float g = bf2f((unsigned short)((r & 1) ? (gpk >> 16) : (gpk & 0xFFFFu)));
      float p = (e / s) * g;
      p += __shfl_xor(p, 1, 32);
      p += __shfl_xor(p, 2, 32);
      p += __shfl_xor(p, 4, 32);
      p += __shfl_xor(p, 8, 32);
      osum[r] = p;
    }
    if (ln == 0) {
#pragma unroll
      for (int r = 0; r < 8; ++r)
        out[((size_t)b * CC + cbase + r) * NN + q] = osum[r];
    }
  }
}

// ---------------------------------------------------------------- launch ----
extern "C" void kernel_launch(void* const* d_in, const int* in_sizes, int n_in,
                              void* d_out, int out_size, void* d_ws,
                              size_t ws_size, hipStream_t stream) {
  (void)in_sizes; (void)n_in; (void)out_size; (void)ws_size;
  const float* pcd   = (const float*)d_in[0];
  const float* feat  = (const float*)d_in[1];
  const float* pcdb  = (const float*)d_in[2];
  const float* featb = (const float*)d_in[3];
  const float* pw1   = (const float*)d_in[4];
  const float* pb1   = (const float*)d_in[5];
  const float* pg    = (const float*)d_in[6];
  const float* pbeta = (const float*)d_in[7];
  const float* pmean = (const float*)d_in[8];
  const float* pvar  = (const float*)d_in[9];
  const float* pw2   = (const float*)d_in[10];
  const float* pb2   = (const float*)d_in[11];
  const float* aw1   = (const float*)d_in[12];
  const float* ab1   = (const float*)d_in[13];
  const float* ag    = (const float*)d_in[14];
  const float* abeta = (const float*)d_in[15];
  const float* amean = (const float*)d_in[16];
  const float* avar  = (const float*)d_in[17];
  const float* aw2   = (const float*)d_in[18];
  const float* ab2   = (const float*)d_in[19];

  // workspace layout (~2.33 MB total)
  char* ws = (char*)d_ws;
  int*            knn  = (int*)(ws);                       // 256 KB
  unsigned short* w2p  = (unsigned short*)(ws + 262144);   // 64 KB
  unsigned short* a1p  = (unsigned short*)(ws + 327680);   // 1 MB
  unsigned short* a2p  = (unsigned short*)(ws + 1376256);  // 1 MB
  float*          ainv = (float*)(ws + 2424832);           // 4 KB
  float*          ashf = (float*)(ws + 2428928);           // 4 KB

  pack_kernel<<<(AH * CC + 255) / 256, 256, 0, stream>>>(
      pw2, aw1, aw2, ag, abeta, amean, avar, w2p, a1p, a2p, ainv, ashf);
  knn_kernel<<<(BB * NN + 255) / 256, 256, 0, stream>>>(pcd, pcdb, knn);
  fused_kernel<<<BB * NN, 256, 0, stream>>>(
      pcd, feat, pcdb, featb, pw1, pb1, pg, pbeta, pmean, pvar, pb2, ab1, ab2,
      w2p, a1p, a2p, ainv, ashf, knn, (float*)d_out);
}
